// GAT_63677185130715
// MI455X (gfx1250) — compile-verified, matrix operand order checked by reading, not run on hardware
//
#include <hip/hip_runtime.h>
#include <math.h>

#define F_IN   512
#define HID    8
#define HEADS  8
#define FEAT1  64      // HEADS*HID
#define NCLS   40
#define NPAD   48      // NCLS padded to 3 WMMA tiles
#define NEG    0.2f
#define KT     32      // WMMA K per step

typedef __attribute__((ext_vector_type(16))) _Float16 v16h;
typedef __attribute__((ext_vector_type(8)))  float    v8f;
typedef __attribute__((ext_vector_type(4)))  int      v4i;

#define AS1 __attribute__((address_space(1)))
#define AS3 __attribute__((address_space(3)))

// ---------------- async staging helpers ----------------

#if defined(__gfx1250__) && __has_builtin(__builtin_amdgcn_global_load_async_to_lds_b128)
#define ASYNC_STAGE 1
#endif

// copy 16 bytes global -> LDS (per calling lane)
__device__ __forceinline__ void stage16(float* lds, const float* g) {
#ifdef ASYNC_STAGE
    __builtin_amdgcn_global_load_async_to_lds_b128(
        (AS1 v4i*)g, (AS3 v4i*)lds, 0, 0);
#else
    *(float4*)lds = *(const float4*)g;
#endif
}

__device__ __forceinline__ void wait_stage() {
#ifdef ASYNC_STAGE
#if __has_builtin(__builtin_amdgcn_s_wait_asynccnt)
    __builtin_amdgcn_s_wait_asynccnt(0);
#else
    asm volatile("s_wait_asynccnt 0x0" ::: "memory");
#endif
#endif
}

// ---------------- misc helpers ----------------

__device__ __forceinline__ void atomicMaxF(float* a, float v) {
    if (v >= 0.0f) atomicMax((int*)a, __float_as_int(v));
    else           atomicMin((unsigned int*)a, __float_as_uint(v));
}

__device__ __forceinline__ void edge_sd(const int* __restrict__ ei, int E, int e,
                                        int& s, int& d) {
    if (e < E) { s = ei[e]; d = ei[E + e]; }
    else       { s = e - E; d = e - E; }          // self-loop
}

__global__ void fill_f32(float* __restrict__ p, float v, int n) {
    int i = blockIdx.x * blockDim.x + threadIdx.x;
    if (i < n) p[i] = v;
}

// ---------------- one-shot weight packing into WMMA B-fragment layout ------
// Bp[((kk*2+hi)*COLS + col)*16 + j] = f16(W[K*WN + col]),
//   K = kk*32 + hi*8 + (j<8 ? j : j+8)   (per 16-bit B-matrix lane layout)

__global__ void pack_w1(const float* __restrict__ W, _Float16* __restrict__ Bp) {
    int idx = blockIdx.x * blockDim.x + threadIdx.x;
    if (idx >= (F_IN / KT) * 2 * FEAT1 * 16) return;      // 32768
    int j   = idx & 15;
    int col = (idx >> 4) & 63;
    int hi  = (idx >> 10) & 1;
    int kk  = idx >> 11;
    int jj  = (j < 8) ? j : j + 8;
    int K   = kk * KT + hi * 8 + jj;
    Bp[idx] = (_Float16)W[K * FEAT1 + col];
}

__global__ void pack_w2(const float* __restrict__ W, _Float16* __restrict__ Bp) {
    int idx = blockIdx.x * blockDim.x + threadIdx.x;
    if (idx >= (FEAT1 / KT) * 2 * NPAD * 16) return;      // 3072
    int j   = idx & 15;
    int col = (idx >> 4) % NPAD;
    int t   = (idx >> 4) / NPAD;                          // kk*2 + hi
    int hi  = t & 1;
    int kk  = t >> 1;
    int jj  = (j < 8) ? j : j + 8;
    int K   = kk * KT + hi * 8 + jj;
    Bp[idx] = (_Float16)((col < NCLS) ? W[K * NCLS + col] : 0.0f);
}

// ---------------- GEMM 1: H[N,64] = X[N,512] @ W1[512,64] ----------------
// fp32 in/out, f16 WMMA. 4 waves/block share one 16-row A tile staged in LDS
// (async double-buffered); wave w owns N-columns [16w,16w+16). N % 16 == 0.
// B fragments are contiguous 32B loads from the pre-packed f16 weights (L2-hot).
__global__ __launch_bounds__(128) void gemm1_wmma(const float* __restrict__ X,
                                                  const _Float16* __restrict__ Bp,
                                                  float* __restrict__ H) {
    __shared__ float sA[2][16 * KT];              // 2 x 2 KB A tiles (fp32)

    int tid = threadIdx.x;
    int mbase = blockIdx.x * 16;
    // staging: thread t copies 16B (row t/8, cols (t%8)*4 .. +3)
    int ar = tid >> 3;
    int ac = (tid & 7) * 4;
    const float* gA = X + (size_t)(mbase + ar) * F_IN + ac;
    float* lA0 = &sA[0][ar * KT + ac];
    float* lA1 = &sA[1][ar * KT + ac];
    stage16(lA0, gA);                              // prefetch tile kk=0

    int wave = tid >> 5, lane = tid & 31;
    int l16 = lane & 15, hi = lane >> 4;           // K-chunk select per ISA layout
    int col = wave * 16 + l16;
    const v16h* bfrag = (const v16h*)Bp + col;     // + (kk*2+hi)*64 per k-step

    v8f acc = {};
    for (int kk = 0; kk < F_IN / KT; ++kk) {
        wait_stage();
        __syncthreads();                           // tile kk ready; tile kk-1 consumed
        if (kk + 1 < F_IN / KT)
            stage16((kk & 1) ? lA0 : lA1, gA + (kk + 1) * KT);

        v16h b = bfrag[(kk * 2 + hi) * FEAT1];     // 32B contiguous per lane
        const float* at = &sA[kk & 1][l16 * KT + hi * 8];
        v16h a;
#pragma unroll
        for (int i = 0; i < 8; ++i) {
            a[i]     = (_Float16)at[i];            // K = kk*32 + hi*8 + i
            a[8 + i] = (_Float16)at[16 + i];       // K = kk*32 + hi*8 + 16 + i
        }
        acc = __builtin_amdgcn_wmma_f32_16x16x32_f16(false, a, false, b,
                                                     (short)0, acc, false, false);
    }
#pragma unroll
    for (int r = 0; r < 8; ++r)
        H[(size_t)(mbase + hi * 8 + r) * FEAT1 + col] = acc[r];
}

// ---------------- GEMM 2: H2[N,40] = He[N,64] @ W2[64,40] ----------------
__global__ __launch_bounds__(128) void gemm2_wmma(const float* __restrict__ He,
                                                  const _Float16* __restrict__ Bp,
                                                  float* __restrict__ H2) {
    __shared__ float sA[16 * FEAT1];               // 4 KB: one 16x64 A tile

    int tid = threadIdx.x;
    int mbase = blockIdx.x * 16;
    // 16x64 fp32 tile is 1024 contiguous floats of He: 128 threads x 8 floats
    const float* gA = He + (size_t)mbase * FEAT1 + tid * 8;
    stage16(&sA[tid * 8],     gA);
    stage16(&sA[tid * 8 + 4], gA + 4);
    wait_stage();
    __syncthreads();

    int wave = tid >> 5, lane = tid & 31;
    if (wave >= 3) return;                         // 3 N-tiles cover 48 cols
    int l16 = lane & 15, hi = lane >> 4;
    int col = wave * 16 + l16;
    const v16h* bfrag = (const v16h*)Bp + col;

    v8f acc = {};
#pragma unroll
    for (int kk = 0; kk < FEAT1 / KT; ++kk) {
        v16h b = bfrag[(kk * 2 + hi) * NPAD];
        const float* at = &sA[l16 * FEAT1 + kk * KT + hi * 8];
        v16h a;
#pragma unroll
        for (int i = 0; i < 8; ++i) {
            a[i]     = (_Float16)at[i];
            a[8 + i] = (_Float16)at[16 + i];
        }
        acc = __builtin_amdgcn_wmma_f32_16x16x32_f16(false, a, false, b,
                                                     (short)0, acc, false, false);
    }
    if (col < NCLS) {                              // lane-uniform over rows
#pragma unroll
        for (int r = 0; r < 8; ++r)
            H2[(size_t)(mbase + hi * 8 + r) * NCLS + col] = acc[r];
    }
}

// ---------------- attention logits ----------------

__global__ void attn_logits1(const float* __restrict__ H, const float* __restrict__ aS,
                             const float* __restrict__ aD, float* __restrict__ lS,
                             float* __restrict__ lD, int N) {
    int i = blockIdx.x * blockDim.x + threadIdx.x;
    if (i >= N * HEADS) return;
    int n = i >> 3, h = i & 7;
    const float* hp = H + (size_t)n * FEAT1 + h * HID;
    const float* as = aS + h * HID;
    const float* ad = aD + h * HID;
    float s = 0.f, d = 0.f;
#pragma unroll
    for (int f = 0; f < HID; ++f) { float v = hp[f]; s += v * as[f]; d += v * ad[f]; }
    lS[i] = s; lD[i] = d;
}

__global__ void attn_logits2(const float* __restrict__ H, const float* __restrict__ aS,
                             const float* __restrict__ aD, float* __restrict__ lS,
                             float* __restrict__ lD, int N) {
    int n = blockIdx.x * blockDim.x + threadIdx.x;
    if (n >= N) return;
    const float* hp = H + (size_t)n * NCLS;
    float s = 0.f, d = 0.f;
#pragma unroll
    for (int c = 0; c < NCLS; ++c) { float v = hp[c]; s += v * aS[c]; d += v * aD[c]; }
    lS[n] = s; lD[n] = d;
}

// ---------------- layer-1 edge passes (per (edge, head)) ----------------

__global__ void edge_max1(const int* __restrict__ ei, int E, int Etot,
                          const float* __restrict__ lS, const float* __restrict__ lD,
                          float* __restrict__ m) {
    int i = blockIdx.x * blockDim.x + threadIdx.x;
    if (i >= Etot * HEADS) return;
    int e = i >> 3, h = i & 7, s, d;
    edge_sd(ei, E, e, s, d);
    float v = lS[s * HEADS + h] + lD[d * HEADS + h];
    v = v > 0.f ? v : NEG * v;
    atomicMaxF(&m[d * HEADS + h], v);
}

__global__ void edge_sum1(const int* __restrict__ ei, int E, int Etot,
                          const float* __restrict__ lS, const float* __restrict__ lD,
                          const float* __restrict__ m, float* __restrict__ dn) {
    int i = blockIdx.x * blockDim.x + threadIdx.x;
    if (i >= Etot * HEADS) return;
    int e = i >> 3, h = i & 7, s, d;
    edge_sd(ei, E, e, s, d);
    float v = lS[s * HEADS + h] + lD[d * HEADS + h];
    v = v > 0.f ? v : NEG * v;
    atomicAdd(&dn[d * HEADS + h], expf(v - m[d * HEADS + h]));
}

__global__ void edge_agg1(const int* __restrict__ ei, int E, int Etot,
                          const float* __restrict__ lS, const float* __restrict__ lD,
                          const float* __restrict__ m, const float* __restrict__ dn,
                          const float* __restrict__ H, float* __restrict__ agg) {
    int i = blockIdx.x * blockDim.x + threadIdx.x;
    if (i >= Etot * HEADS) return;
    int e = i >> 3, h = i & 7, s, d;
    edge_sd(ei, E, e, s, d);
    const float* hp = H + (size_t)s * FEAT1 + h * HID;
    __builtin_prefetch(hp, 0, 0);                  // global_prefetch of gathered row
    float v = lS[s * HEADS + h] + lD[d * HEADS + h];
    v = v > 0.f ? v : NEG * v;
    float alpha = expf(v - m[d * HEADS + h]) / dn[d * HEADS + h];
    float* op = agg + (size_t)d * FEAT1 + h * HID;
#pragma unroll
    for (int f = 0; f < HID; ++f) atomicAdd(&op[f], alpha * hp[f]);
}

// ---------------- layer-2 edge passes (1 head, 40 channels) ----------------

__global__ void edge_max2(const int* __restrict__ ei, int E, int Etot,
                          const float* __restrict__ lS, const float* __restrict__ lD,
                          float* __restrict__ m) {
    int e = blockIdx.x * blockDim.x + threadIdx.x;
    if (e >= Etot) return;
    int s, d; edge_sd(ei, E, e, s, d);
    float v = lS[s] + lD[d];
    v = v > 0.f ? v : NEG * v;
    atomicMaxF(&m[d], v);
}

__global__ void edge_sum2(const int* __restrict__ ei, int E, int Etot,
                          const float* __restrict__ lS, const float* __restrict__ lD,
                          const float* __restrict__ m, float* __restrict__ dn) {
    int e = blockIdx.x * blockDim.x + threadIdx.x;
    if (e >= Etot) return;
    int s, d; edge_sd(ei, E, e, s, d);
    float v = lS[s] + lD[d];
    v = v > 0.f ? v : NEG * v;
    atomicAdd(&dn[d], expf(v - m[d]));
}

__global__ void edge_agg2(const int* __restrict__ ei, int E, int Etot,
                          const float* __restrict__ lS, const float* __restrict__ lD,
                          const float* __restrict__ m, const float* __restrict__ dn,
                          const float* __restrict__ H, float* __restrict__ agg) {
    int i = blockIdx.x * blockDim.x + threadIdx.x;
    if (i >= Etot * 8) return;                     // 8 threads/edge, 5 feats each
    int e = i >> 3, part = i & 7, s, d;
    edge_sd(ei, E, e, s, d);
    int f0 = part * 5;
    const float* hp = H + (size_t)s * NCLS + f0;
    __builtin_prefetch(hp, 0, 0);
    float v = lS[s] + lD[d];
    v = v > 0.f ? v : NEG * v;
    float alpha = expf(v - m[d]) / dn[d];
    float* op = agg + (size_t)d * NCLS + f0;
#pragma unroll
    for (int f = 0; f < 5; ++f) atomicAdd(&op[f], alpha * hp[f]);
}

// ---------------- elementwise epilogues ----------------

__global__ void bias_elu(const float* __restrict__ agg, const float* __restrict__ b,
                         float* __restrict__ out, int total) {
    int i = blockIdx.x * blockDim.x + threadIdx.x;
    if (i >= total) return;
    float v = agg[i] + b[i % FEAT1];
    out[i] = v > 0.f ? v : (expf(v) - 1.0f);
}

__global__ void bias_logsoftmax(const float* __restrict__ agg, const float* __restrict__ b,
                                float* __restrict__ out, int N) {
    int n = blockIdx.x * blockDim.x + threadIdx.x;
    if (n >= N) return;
    float z[NCLS];
    float mx = -INFINITY;
#pragma unroll
    for (int c = 0; c < NCLS; ++c) {
        z[c] = agg[(size_t)n * NCLS + c] + b[c];
        mx = fmaxf(mx, z[c]);
    }
    float se = 0.f;
#pragma unroll
    for (int c = 0; c < NCLS; ++c) se += expf(z[c] - mx);
    float ls = mx + logf(se);
#pragma unroll
    for (int c = 0; c < NCLS; ++c) out[(size_t)n * NCLS + c] = z[c] - ls;
}

// ---------------- launch ----------------

extern "C" void kernel_launch(void* const* d_in, const int* in_sizes, int n_in,
                              void* d_out, int out_size, void* d_ws, size_t ws_size,
                              hipStream_t stream) {
    const float* x   = (const float*)d_in[0];
    const int*   ei  = (const int*)  d_in[1];
    const float* W1  = (const float*)d_in[2];
    const float* aS1 = (const float*)d_in[3];
    const float* aD1 = (const float*)d_in[4];
    const float* b1  = (const float*)d_in[5];
    const float* W2  = (const float*)d_in[6];
    const float* aS2 = (const float*)d_in[7];
    const float* aD2 = (const float*)d_in[8];
    const float* b2  = (const float*)d_in[9];
    float* out = (float*)d_out;

    const int N    = in_sizes[0] / F_IN;   // 100000 (divisible by 16)
    const int E    = in_sizes[1] / 2;      // 1600000
    const int Etot = E + N;                // + self loops

    // workspace layout (floats), with reuse
    float* ws   = (float*)d_ws;
    float* h1   = ws;                              // N*64 (later: ELU features)
    float* agg1 = h1   + (size_t)N * FEAT1;        // N*64 (later reused as agg2)
    float* h2   = agg1 + (size_t)N * FEAT1;        // N*40
    float* lS1  = h2   + (size_t)N * NCLS;         // N*8
    float* lD1  = lS1  + (size_t)N * HEADS;        // N*8
    float* m1   = lD1  + (size_t)N * HEADS;        // N*8
    float* dn1  = m1   + (size_t)N * HEADS;        // N*8
    float* lS2  = lS1;                             // N   (reused after layer 1)
    float* lD2  = lS1 + N;
    float* m2   = lS1 + 2 * N;
    float* dn2  = lS1 + 3 * N;
    float* agg2 = agg1;

    // packed f16 weights at the tail (32B-aligned)
    size_t tail = (((size_t)N * 200 * sizeof(float)) + 255) & ~(size_t)255;
    _Float16* Bp1 = (_Float16*)((char*)d_ws + tail);               // 32768 halves
    _Float16* Bp2 = Bp1 + (F_IN / KT) * 2 * FEAT1 * 16;            // 3072 halves

    auto cdiv = [](long long a, int b) { return (unsigned)((a + b - 1) / b); };
    const int T = 256;

    // ---- weight packing (tiny, once per launch) ----
    pack_w1<<<cdiv((F_IN / KT) * 2 * FEAT1 * 16, T), T, 0, stream>>>(W1, Bp1);
    pack_w2<<<cdiv((FEAT1 / KT) * 2 * NPAD * 16, T), T, 0, stream>>>(W2, Bp2);

    // ---- layer 1 ----
    gemm1_wmma<<<N / 16, 128, 0, stream>>>(x, Bp1, h1);
    attn_logits1<<<cdiv((long long)N * HEADS, T), T, 0, stream>>>(h1, aS1, aD1, lS1, lD1, N);
    fill_f32<<<cdiv((long long)N * FEAT1, T), T, 0, stream>>>(agg1, 0.f, N * FEAT1);
    fill_f32<<<cdiv((long long)N * HEADS, T), T, 0, stream>>>(m1, -INFINITY, N * HEADS);
    fill_f32<<<cdiv((long long)N * HEADS, T), T, 0, stream>>>(dn1, 0.f, N * HEADS);
    edge_max1<<<cdiv((long long)Etot * HEADS, T), T, 0, stream>>>(ei, E, Etot, lS1, lD1, m1);
    edge_sum1<<<cdiv((long long)Etot * HEADS, T), T, 0, stream>>>(ei, E, Etot, lS1, lD1, m1, dn1);
    edge_agg1<<<cdiv((long long)Etot * HEADS, T), T, 0, stream>>>(ei, E, Etot, lS1, lD1, m1, dn1, h1, agg1);
    bias_elu<<<cdiv((long long)N * FEAT1, T), T, 0, stream>>>(agg1, b1, h1, N * FEAT1);

    // ---- layer 2 ----
    gemm2_wmma<<<N / 16, 128, 0, stream>>>(h1, Bp2, h2);
    attn_logits2<<<cdiv(N, T), T, 0, stream>>>(h2, aS2, aD2, lS2, lD2, N);
    fill_f32<<<cdiv((long long)N * NCLS, T), T, 0, stream>>>(agg2, 0.f, N * NCLS);
    fill_f32<<<cdiv(N, T), T, 0, stream>>>(m2, -INFINITY, N);
    fill_f32<<<cdiv(N, T), T, 0, stream>>>(dn2, 0.f, N);
    edge_max2<<<cdiv(Etot, T), T, 0, stream>>>(ei, E, Etot, lS2, lD2, m2);
    edge_sum2<<<cdiv(Etot, T), T, 0, stream>>>(ei, E, Etot, lS2, lD2, m2, dn2);
    edge_agg2<<<cdiv((long long)Etot * 8, T), T, 0, stream>>>(ei, E, Etot, lS2, lD2, m2, dn2, h2, agg2);
    bias_logsoftmax<<<cdiv(N, T), T, 0, stream>>>(agg2, b2, out, N);
}
